// MultiHeadAttention_27444841022070
// MI455X (gfx1250) — matrix operand order repaired and checked by reading, not compiled, not run on hardware
//
#include <hip/hip_runtime.h>

#define S 4096
#define D 768
#define H 12
#define DK 64
#define ST (S/16)   // 256 row tiles
#define DT (D/16)   // 48 col tiles
#define KT (D/32)   // 24 k tiles
#define AW 8        // waves per attention workgroup

typedef __attribute__((ext_vector_type(16))) __bf16 v16bf;
typedef __attribute__((ext_vector_type(8)))  float  v8f;
typedef unsigned short u16;
typedef unsigned int   u32;
typedef unsigned long long u64;

__device__ __forceinline__ v8f wmma_bf16(v16bf a, v16bf b, v8f c) {
  return __builtin_amdgcn_wmma_f32_16x16x32_bf16(false, a, false, b, (short)0, c, false, false);
}

__device__ __forceinline__ u16 f2bf(float f) {
  u32 u = __float_as_uint(f);
  u32 r = u + 0x7FFFu + ((u >> 16) & 1u);   // round-to-nearest-even
  return (u16)(r >> 16);
}

// Low 32 bits of a flat pointer to __shared__ == LDS byte offset (ISA 10.2:
// the LDS address is the low 32 address bits).
__device__ __forceinline__ u32 lds_off(const void* p) {
  return (u32)(uintptr_t)p;
}

// One async 16B-per-lane copy: global -> LDS, ASYNCcnt-tracked.
__device__ __forceinline__ void async_cp_b128(u32 ldsa, const void* gp) {
  u64 ga = (u64)(uintptr_t)gp;
  asm volatile("global_load_async_to_lds_b128 %0, %1, off"
               :: "v"(ldsa), "v"(ga) : "memory");
}

// A-fragment (16x32 bf16) per-lane layout:
// lane l (m = l%16), element e: kk = (l<16 ? 0 : 16) + e
__device__ __forceinline__ void inv_a(int m, int kk, int& lane, int& e) {
  lane = m + (kk < 16 ? 0 : 16); e = kk & 15;
}
// B-fragment (32x16 bf16): lane l (n = l%16), element e: kk = (l<16?0:16)+e

// fp32 [rows,cols] -> bf16 A-frag tiles [rows/16][cols/32][32 lanes][16 elems]
__global__ void pack_a_kernel(const float* __restrict__ x, u16* __restrict__ out,
                              int rows, int cols) {
  int idx = blockIdx.x * blockDim.x + threadIdx.x;
  if (idx >= rows * cols) return;
  int tile = idx >> 9, within = idx & 511;
  int lane = within >> 4, e = within & 15;
  int ctiles = cols >> 5;
  int mt = tile / ctiles, kt = tile % ctiles;
  int m  = lane & 15;
  int kk = (lane < 16 ? 0 : 16) + e;
  out[idx] = f2bf(x[(size_t)(mt * 16 + m) * cols + kt * 32 + kk]);
}

// fp32 [K,N] -> bf16 B-frag tiles [K/32][N/16][32][16]
__global__ void pack_b_kernel(const float* __restrict__ w, u16* __restrict__ out,
                              int krows, int ncols) {
  int idx = blockIdx.x * blockDim.x + threadIdx.x;
  if (idx >= krows * ncols) return;
  int tile = idx >> 9, within = idx & 511;
  int lane = within >> 4, e = within & 15;
  int ntiles = ncols >> 4;
  int kt = tile / ntiles, nt = tile % ntiles;
  int n  = lane & 15;
  int kk = (lane < 16 ? 0 : 16) + e;
  out[idx] = f2bf(w[(size_t)(kt * 32 + kk) * ncols + nt * 16 + n]);
}

// int32 mask [S,S] -> bit-packed [S][S/32]
__global__ void pack_mask_kernel(const int* __restrict__ mask, u32* __restrict__ bits) {
  int w = blockIdx.x * blockDim.x + threadIdx.x;
  if (w >= (S * S) / 32) return;
  const int* p = mask + (size_t)w * 32;
  u32 b = 0;
#pragma unroll
  for (int i = 0; i < 32; ++i) b |= (p[i] != 0 ? 1u : 0u) << i;
  bits[w] = b;
}

// GEMM: X(S x D, A-frag) * W(D x D, B-frag) + bias.
// One wave computes a 64x64 output block (4x4 WMMA tiles, register-blocked:
// 16 WMMAs per K-step off 4 A-frags + 4 B-frags -> 0.5KB of L2 traffic/WMMA).
// ~224 VGPRs/wave -> launch_bounds(128,2) gives a ~512-VGPR budget (no spills).
// mode 0: scatter -> Q A-frag packed [H][ST][2][512]
// mode 1: scatter -> K^T B-frag packed [H][ST(key tiles)][2][512]
// mode 2: scatter -> V B-frag packed [H][S/32][4][512]
// mode 3: fp32 row-major output [S][D]
__global__ void __launch_bounds__(128, 2)
proj_kernel(const u16* __restrict__ xA, const u16* __restrict__ wB,
            const float* __restrict__ bias, u16* __restrict__ outPk,
            float* __restrict__ outF, int mode) {
  int lane = threadIdx.x & 31;
  int wave = threadIdx.x >> 5;
  const int NT64 = DT / 4;                    // 12
  int t64 = blockIdx.x * 4 + wave;            // 0 .. (S/64)*(D/64)-1 = 767
  int mt64 = t64 / NT64, nt64 = t64 % NT64;

  v8f acc[4][4];
  v8f zero = {0.f, 0.f, 0.f, 0.f, 0.f, 0.f, 0.f, 0.f};
#pragma unroll
  for (int i = 0; i < 4; ++i)
#pragma unroll
    for (int j = 0; j < 4; ++j) acc[i][j] = zero;

  const u16* xbase = xA + (size_t)(mt64 * 4 * KT) * 512 + lane * 16;
  const u16* wbase = wB + (size_t)(nt64 * 4) * 512 + lane * 16;
  for (int kt = 0; kt < KT; ++kt) {
    v16bf a[4], b[4];
#pragma unroll
    for (int i = 0; i < 4; ++i)
      a[i] = *(const v16bf*)(xbase + (size_t)(i * KT + kt) * 512);
#pragma unroll
    for (int j = 0; j < 4; ++j)
      b[j] = *(const v16bf*)(wbase + (size_t)(kt * DT + j) * 512);
#pragma unroll
    for (int i = 0; i < 4; ++i)
#pragma unroll
      for (int j = 0; j < 4; ++j)
        acc[i][j] = wmma_bf16(a[i], b[j], acc[i][j]);
  }

  int half = lane >> 4;
#pragma unroll
  for (int i = 0; i < 4; ++i) {
#pragma unroll
    for (int j = 0; j < 4; ++j) {
      int mt = mt64 * 4 + i, nt = nt64 * 4 + j;
      int n0 = nt * 16 + (lane & 15);
      float bn = bias[n0];
      int h = n0 / DK, dk = n0 % DK;
#pragma unroll
      for (int r = 0; r < 8; ++r) {
        float v = acc[i][j][r] + bn;
        int row  = r + half * 8;       // local m in tile (C-layout)
        int srow = mt * 16 + row;      // global row
        if (mode == 3) {
          outF[(size_t)srow * D + n0] = v;
        } else if (mode == 0) {
          int kt2 = dk >> 5, kk = dk & 31;
          int ld, ed; inv_a(row, kk, ld, ed);
          outPk[((size_t)((h * ST + mt) * 2 + kt2)) * 512 + ld * 16 + ed] = f2bf(v);
        } else if (mode == 1) {
          int kt2 = dk >> 5, kk = dk & 31;
          int ld = (srow & 15) + (kk < 16 ? 0 : 16);
          int ed = kk & 15;
          int ntk = srow >> 4;
          outPk[((size_t)((h * ST + ntk) * 2 + kt2)) * 512 + ld * 16 + ed] = f2bf(v);
        } else {
          int ntile = dk >> 4, nn = dk & 15;
          int ktv = srow >> 5, kkv = srow & 31;
          int ld = nn + (kkv < 16 ? 0 : 16);
          int ed = kkv & 15;
          outPk[((size_t)((h * (S / 32) + ktv) * 4 + ntile)) * 512 + ld * 16 + ed] = f2bf(v);
        }
      }
    }
  }
}

// Flash attention: 8 waves/workgroup, one head, 8 query tiles. K/V key blocks
// (8KB) staged cooperatively into double-buffered LDS with async copies.
__global__ void attn_kernel(const u16* __restrict__ qA, const u16* __restrict__ kB,
                            const u16* __restrict__ vB, const u32* __restrict__ mbits,
                            u16* __restrict__ ctxA) {
  __shared__ __align__(16) u32 s_mask[AW][16 * 128];   // 64 KB mask bits
  __shared__ __align__(16) u16 s_p[AW][512];           // 8 KB P scratch
  __shared__ __align__(16) u16 s_kv[2][4096];          // 16 KB: [K 2048 | V 2048] x2

  int lane = threadIdx.x & 31;
  int wave = threadIdx.x >> 5;
  int qt = blockIdx.x * AW + wave;               // query tile (16 rows)
  int h  = blockIdx.y;

  // async-stage this wave's mask rows: 8KB contiguous -> s_mask[wave]
  {
    const u32* msrc = mbits + (size_t)(qt * 16) * 128;
    u32 mdst = lds_off(&s_mask[wave][0]) + (u32)lane * 16;
#pragma unroll
    for (int i = 0; i < 16; ++i)
      async_cp_b128(mdst + i * 512, msrc + i * 128 + lane * 4);
  }
  // async-stage K/V block 0 into buffer 0: wave w copies 1KB of the 8KB
  {
    const u16* ks = kB + (size_t)(h * ST) * 1024;          // block 0 K (4KB)
    const u16* vs = vB + (size_t)(h * (S / 32)) * 2048;    // block 0 V (4KB)
    const u16* src = (wave < 4) ? (ks + wave * 512) : (vs + (wave - 4) * 512);
    u32 dst = lds_off(&s_kv[0][0]) + (u32)wave * 1024 + (u32)lane * 16;
    async_cp_b128(dst,       src + lane * 8);
    async_cp_b128(dst + 512, src + 256 + lane * 8);
  }

  const u16* qb = qA + ((size_t)(h * ST + qt) * 2) * 512 + lane * 16;
  v16bf aq0 = *(const v16bf*)qb;
  v16bf aq1 = *(const v16bf*)(qb + 512);

  v8f acc0 = {0.f,0.f,0.f,0.f,0.f,0.f,0.f,0.f};
  v8f acc1 = acc0, acc2 = acc0, acc3 = acc0;
  float mrun[8], lrun[8];
#pragma unroll
  for (int r = 0; r < 8; ++r) { mrun[r] = -__builtin_inff(); lrun[r] = 0.f; }

  int half = lane >> 4;
  int col  = lane & 15;

  for (int kb = 0; kb < S / 32; ++kb) {
    int par = kb & 1;
    if (kb + 1 < S / 32) {   // issue async copy of next block into other buffer
      const u16* ks = kB + (size_t)(h * ST + 2 * (kb + 1)) * 1024;
      const u16* vs = vB + (size_t)(h * (S / 32) + (kb + 1)) * 2048;
      const u16* src = (wave < 4) ? (ks + wave * 512) : (vs + (wave - 4) * 512);
      u32 dst = lds_off(&s_kv[par ^ 1][0]) + (u32)wave * 1024 + (u32)lane * 16;
      async_cp_b128(dst,       src + lane * 8);
      async_cp_b128(dst + 512, src + 256 + lane * 8);
      asm volatile("s_wait_asynccnt 0x2" ::: "memory");  // drain all but the 2 just issued
    } else {
      asm volatile("s_wait_asynccnt 0x0" ::: "memory");
    }
    __syncthreads();   // all waves' copies for buf[par] visible workgroup-wide

    const u16* kp = &s_kv[par][0]    + lane * 16;   // K^T B-frags
    const u16* vp = &s_kv[par][2048] + lane * 16;   // V   B-frags
    v16bf b00 = *(const v16bf*)kp;
    v16bf b01 = *(const v16bf*)(kp + 512);
    v16bf b10 = *(const v16bf*)(kp + 1024);
    v16bf b11 = *(const v16bf*)(kp + 1536);
    v8f s0 = {0.f,0.f,0.f,0.f,0.f,0.f,0.f,0.f};
    v8f s1 = s0;
    s0 = wmma_bf16(aq0, b00, s0);
    s0 = wmma_bf16(aq1, b01, s0);
    s1 = wmma_bf16(aq0, b10, s1);
    s1 = wmma_bf16(aq1, b11, s1);

#pragma unroll
    for (int r = 0; r < 8; ++r) {
      int row = r + half * 8;
      u32 word = s_mask[wave][row * 128 + kb];
      float x0 = ((word >> col)        & 1u) ? s0[r] * 0.125f : -1e9f;
      float x1 = ((word >> (16 + col)) & 1u) ? s1[r] * 0.125f : -1e9f;
      float mx = fmaxf(x0, x1);
      mx = fmaxf(mx, __shfl_xor(mx, 1, 32));
      mx = fmaxf(mx, __shfl_xor(mx, 2, 32));
      mx = fmaxf(mx, __shfl_xor(mx, 4, 32));
      mx = fmaxf(mx, __shfl_xor(mx, 8, 32));
      float newm = fmaxf(mrun[r], mx);
      float corr = __expf(mrun[r] - newm);
      float p0 = __expf(x0 - newm);
      float p1 = __expf(x1 - newm);
      float rs = p0 + p1;
      rs += __shfl_xor(rs, 1, 32);
      rs += __shfl_xor(rs, 2, 32);
      rs += __shfl_xor(rs, 4, 32);
      rs += __shfl_xor(rs, 8, 32);
      mrun[r] = newm;
      lrun[r] = lrun[r] * corr + rs;
      acc0[r] *= corr; acc1[r] *= corr; acc2[r] *= corr; acc3[r] *= corr;
      int ld, ed;
      inv_a(row, col,      ld, ed); s_p[wave][ld * 16 + ed] = f2bf(p0);
      inv_a(row, 16 + col, ld, ed); s_p[wave][ld * 16 + ed] = f2bf(p1);
    }
    asm volatile("s_wait_dscnt 0x0" ::: "memory");
    v16bf ap = *(const v16bf*)(&s_p[wave][lane * 16]);
    acc0 = wmma_bf16(ap, *(const v16bf*)vp,          acc0);
    acc1 = wmma_bf16(ap, *(const v16bf*)(vp +  512), acc1);
    acc2 = wmma_bf16(ap, *(const v16bf*)(vp + 1024), acc2);
    acc3 = wmma_bf16(ap, *(const v16bf*)(vp + 1536), acc3);

    __syncthreads();   // nobody overwrites buf[par] until all waves finished it
  }

#pragma unroll
  for (int r = 0; r < 8; ++r) {
    int row = r + half * 8;
    float inv = 1.0f / lrun[r];
#pragma unroll
    for (int nt = 0; nt < 4; ++nt) {
      float v = (nt == 0 ? acc0[r] : nt == 1 ? acc1[r] : nt == 2 ? acc2[r] : acc3[r]) * inv;
      int d = h * 64 + nt * 16 + col;
      int kt2 = d >> 5, kk = d & 31;
      int ld, ed; inv_a(row, kk, ld, ed);
      ctxA[((size_t)(qt * KT + kt2)) * 512 + ld * 16 + ed] = f2bf(v);
    }
  }
}

extern "C" void kernel_launch(void* const* d_in, const int* in_sizes, int n_in,
                              void* d_out, int out_size, void* d_ws, size_t ws_size,
                              hipStream_t stream) {
  const float* q    = (const float*)d_in[0];
  const float* k    = (const float*)d_in[1];
  const float* v    = (const float*)d_in[2];
  const int*   mask = (const int*)  d_in[3];
  const float* wq   = (const float*)d_in[4];
  const float* bq   = (const float*)d_in[5];
  const float* wk   = (const float*)d_in[6];
  const float* bk   = (const float*)d_in[7];
  const float* wv   = (const float*)d_in[8];
  const float* bv   = (const float*)d_in[9];
  const float* wo   = (const float*)d_in[10];
  const float* bo   = (const float*)d_in[11];

  char* ws = (char*)d_ws;
  const size_t SD2 = (size_t)S * D * 2;   // 6 MB
  const size_t DD2 = (size_t)D * D * 2;   // 1.125 MB
  u16* qIn  = (u16*)(ws);
  u16* kIn  = (u16*)(ws + 1 * SD2);
  u16* vIn  = (u16*)(ws + 2 * SD2);
  u16* wqB  = (u16*)(ws + 3 * SD2);
  u16* wkB  = (u16*)(ws + 3 * SD2 + 1 * DD2);
  u16* wvB  = (u16*)(ws + 3 * SD2 + 2 * DD2);
  u16* woB  = (u16*)(ws + 3 * SD2 + 3 * DD2);
  u16* qAp  = (u16*)(ws + 3 * SD2 + 4 * DD2);
  u16* kAp  = (u16*)(ws + 4 * SD2 + 4 * DD2);
  u16* vAp  = (u16*)(ws + 5 * SD2 + 4 * DD2);
  u16* ctxA = (u16*)(ws + 6 * SD2 + 4 * DD2);
  u32* mb   = (u32*)(ws + 7 * SD2 + 4 * DD2);

  pack_a_kernel<<<(S * D) / 256, 256, 0, stream>>>(q, qIn, S, D);
  pack_a_kernel<<<(S * D) / 256, 256, 0, stream>>>(k, kIn, S, D);
  pack_a_kernel<<<(S * D) / 256, 256, 0, stream>>>(v, vIn, S, D);
  pack_b_kernel<<<(D * D) / 256, 256, 0, stream>>>(wq, wqB, D, D);
  pack_b_kernel<<<(D * D) / 256, 256, 0, stream>>>(wk, wkB, D, D);
  pack_b_kernel<<<(D * D) / 256, 256, 0, stream>>>(wv, wvB, D, D);
  pack_b_kernel<<<(D * D) / 256, 256, 0, stream>>>(wo, woB, D, D);
  pack_mask_kernel<<<(S * S / 32) / 256, 256, 0, stream>>>(mask, mb);

  const int PROJ_BLOCKS = (S / 64) * (D / 64) / 4;   // 192 blocks of 4 waves
  proj_kernel<<<PROJ_BLOCKS, 128, 0, stream>>>(qIn, wqB, bq, qAp, nullptr, 0);
  proj_kernel<<<PROJ_BLOCKS, 128, 0, stream>>>(kIn, wkB, bk, kAp, nullptr, 1);
  proj_kernel<<<PROJ_BLOCKS, 128, 0, stream>>>(vIn, wvB, bv, vAp, nullptr, 2);

  attn_kernel<<<dim3(ST / AW, H), 256, 0, stream>>>(qAp, kAp, vAp, mb, ctxA);

  proj_kernel<<<PROJ_BLOCKS, 128, 0, stream>>>(ctxA, woB, bo, nullptr, (float*)d_out, 3);
}
